// MultiheadSelfAttention_36283883716657
// MI455X (gfx1250) — compile-verified, matrix operand order checked by reading, not run on hardware
//
#include <hip/hip_runtime.h>
#include <math.h>

#define B_  2
#define S_  2048
#define D_  1024
#define H_  16
#define DK_ 64

typedef _Float16 v16h __attribute__((ext_vector_type(16)));
typedef _Float16 v8h  __attribute__((ext_vector_type(8)));
typedef float    v8f  __attribute__((ext_vector_type(8)));

__device__ __forceinline__ v8f zero_v8f() {
  v8f z;
#pragma unroll
  for (int i = 0; i < 8; ++i) z[i] = 0.0f;
  return z;
}

// 16-bit WMMA operand fragment (16x32, row-major source, this lane's row ptr).
// Lane 0-15: K = {0..7, 16..23}; lane 16-31: K = {8..15, 24..31} (half8 = 0/8).
__device__ __forceinline__ v16h load_frag(const _Float16* rowptr, int half8) {
  v8h lo = *(const v8h*)(rowptr + half8);
  v8h hi = *(const v8h*)(rowptr + 16 + half8);
  v16h r;
#pragma unroll
  for (int i = 0; i < 8; ++i) { r[i] = lo[i]; r[i + 8] = hi[i]; }
  return r;
}

__global__ void f32_to_f16(const float* __restrict__ src, _Float16* __restrict__ dst, size_t n) {
  size_t i = (size_t)blockIdx.x * blockDim.x + threadIdx.x;
  if (i < n) dst[i] = (_Float16)src[i];
}

// C[M,N] (f32) = A[M,K] (f16, row-major) x W[N,K]^T (f16, row-major over K).
// One wave -> 32x64 output tile: 2 A-frags x 4 W-frags = 8 WMMA per 32-wide K step.
__global__ __launch_bounds__(256) void gemm_f16_wmma(
    const _Float16* __restrict__ A, const _Float16* __restrict__ W,
    float* __restrict__ C, int M, int N, int K, int totalTiles) {
  const int lane  = threadIdx.x & 31;
  const int wave  = threadIdx.x >> 5;
  const int id    = blockIdx.x * 8 + wave;
  if (id >= totalTiles) return;
  const int tilesN = N >> 6;
  const int tm = id / tilesN;      // 32-row tile index
  const int tn = id % tilesN;      // 64-col tile index
  const int row   = lane & 15;
  const int half8 = (lane >> 4) << 3;

  const _Float16* Arow0 = A + (size_t)(tm * 32 + row) * K;
  const _Float16* Arow1 = Arow0 + (size_t)16 * K;
  const _Float16* Wrow  = W + (size_t)(tn * 64 + row) * K;

  v8f acc[2][4];
#pragma unroll
  for (int m = 0; m < 2; ++m)
#pragma unroll
    for (int t = 0; t < 4; ++t) acc[m][t] = zero_v8f();

  for (int k = 0; k < K; k += 32) {
    v16h af0 = load_frag(Arow0 + k, half8);
    v16h af1 = load_frag(Arow1 + k, half8);
#pragma unroll
    for (int t = 0; t < 4; ++t) {
      v16h wf = load_frag(Wrow + (size_t)t * 16 * K + k, half8);
      acc[0][t] = __builtin_amdgcn_wmma_f32_16x16x32_f16(
          false, af0, false, wf, (short)0, acc[0][t], false, false);
      acc[1][t] = __builtin_amdgcn_wmma_f32_16x16x32_f16(
          false, af1, false, wf, (short)0, acc[1][t], false, false);
    }
  }

  const int n0 = tn * 64 + row;
#pragma unroll
  for (int m = 0; m < 2; ++m)
#pragma unroll
    for (int t = 0; t < 4; ++t)
#pragma unroll
      for (int j = 0; j < 8; ++j)
        C[(size_t)(tm * 32 + m * 16 + half8 + j) * N + n0 + t * 16] = acc[m][t][j];
}

// RoPE + split QKV (f32, rows = (b,s), cols = (q,h,dk)) into per-(b,h) buffers:
//   Q [S,DK] f16 (pre-scaled by 1/sqrt(DK)), K [S,DK] f16, V^T [DK,S] f16.
// V is stored transposed so the attention P.V B-operand loads are contiguous.
__global__ void rope_split(const float* __restrict__ qkv, const int* __restrict__ tokpos,
                           _Float16* __restrict__ qf, _Float16* __restrict__ kf,
                           _Float16* __restrict__ vt) {
  const size_t total = (size_t)B_ * H_ * S_ * (DK_ / 2);
  const size_t i = (size_t)blockIdx.x * blockDim.x + threadIdx.x;
  if (i >= total) return;
  const int    p  = (int)(i & (DK_ / 2 - 1));
  const size_t t  = i >> 5;                 // DK_/2 == 32
  const int    s  = (int)(t % S_);
  const int    bh = (int)(t / S_);
  const int    h  = bh % H_;
  const int    b  = bh / H_;

  const size_t rowbase = ((size_t)b * S_ + s) * (3 * H_ * DK_);
  const int    col = 2 * p;
  const size_t qo = rowbase + (0 * H_ + h) * DK_ + col;
  const size_t ko = rowbase + (1 * H_ + h) * DK_ + col;
  const size_t vo = rowbase + (2 * H_ + h) * DK_ + col;

  const float x0 = qkv[qo], x1 = qkv[qo + 1];
  const float k0 = qkv[ko], k1 = qkv[ko + 1];
  const float v0 = qkv[vo], v1 = qkv[vo + 1];

  const float pos  = (float)tokpos[s];
  const float freq = __powf(10000.0f, -2.0f * (float)p / (float)DK_);
  float sth, cth;
  __sincosf(pos * freq, &sth, &cth);

  const float qs = 0.125f;  // 1/sqrt(64) folded into Q
  const size_t ob = ((size_t)bh * S_ + s) * DK_ + col;
  qf[ob]     = (_Float16)((x0 * cth - x1 * sth) * qs);
  qf[ob + 1] = (_Float16)((x1 * cth + x0 * sth) * qs);
  kf[ob]     = (_Float16)(k0 * cth - k1 * sth);
  kf[ob + 1] = (_Float16)(k1 * cth + k0 * sth);
  // transposed V: vt[bh][dk][s]
  const size_t vb = ((size_t)bh * DK_ + col) * S_ + s;
  vt[vb]      = (_Float16)v0;
  vt[vb + S_] = (_Float16)v1;
}

// Causal flash attention, one wave per 16-row Q tile, KV streamed 32 at a time.
// 4 score WMMA + 4 PV WMMA per KV step; online softmax via wave32 shuffles.
__global__ __launch_bounds__(256) void attn_wmma(
    const _Float16* __restrict__ Qg, const _Float16* __restrict__ Kg,
    const _Float16* __restrict__ Vtg, _Float16* __restrict__ Og) {
  __shared__ _Float16 plds[8][16 * 32];   // per-wave P-tile staging (1 KB each)

  const int lane  = threadIdx.x & 31;
  const int wave  = threadIdx.x >> 5;
  const int tile  = blockIdx.x * 8 + wave;   // 0 .. B_*H_*(S_/16)-1
  const int qt    = tile & 127;              // S_/16 == 128
  const int bh    = tile >> 7;
  const int q0    = qt * 16;
  const int row   = lane & 15;
  const int half8 = (lane >> 4) << 3;

  const _Float16* Q  = Qg  + (size_t)bh * S_ * DK_;
  const _Float16* K  = Kg  + (size_t)bh * S_ * DK_;
  const _Float16* Vt = Vtg + (size_t)bh * DK_ * S_;   // [DK][S]
  _Float16* my = &plds[wave][0];

  v16h qf[2];
#pragma unroll
  for (int c = 0; c < 2; ++c)
    qf[c] = load_frag(Q + (size_t)(q0 + row) * DK_ + c * 32, half8);

  v8f o[4];
#pragma unroll
  for (int t = 0; t < 4; ++t) o[t] = zero_v8f();
  float rl[8], rm[8];
#pragma unroll
  for (int j = 0; j < 8; ++j) { rl[j] = 0.0f; rm[j] = -__builtin_inff(); }

  for (int kv0 = 0; kv0 <= q0 + 15; kv0 += 32) {
    // ---- scores: Q(16x64) . K(32x64)^T -> two 16x16 f32 tiles ----
    v8f s0 = zero_v8f(), s1 = zero_v8f();
#pragma unroll
    for (int c = 0; c < 2; ++c) {
      v16h kf0 = load_frag(K + (size_t)(kv0 + row) * DK_ + c * 32, half8);
      s0 = __builtin_amdgcn_wmma_f32_16x16x32_f16(false, qf[c], false, kf0, (short)0, s0, false, false);
      v16h kf1 = load_frag(K + (size_t)(kv0 + 16 + row) * DK_ + c * 32, half8);
      s1 = __builtin_amdgcn_wmma_f32_16x16x32_f16(false, qf[c], false, kf1, (short)0, s1, false, false);
    }

    // ---- causal mask (only tiles touching the diagonal) ----
    if (kv0 + 31 > q0) {
#pragma unroll
      for (int j = 0; j < 8; ++j) {
        const int qg = q0 + half8 + j;
        if (kv0 + row > qg)      s0[j] = -1e30f;
        if (kv0 + 16 + row > qg) s1[j] = -1e30f;
      }
    }

    // ---- online softmax; rows live in (vgpr j, lane-half), cols across 16 lanes ----
#pragma unroll
    for (int j = 0; j < 8; ++j) {
      float v = fmaxf(s0[j], s1[j]);
#pragma unroll
      for (int off = 1; off < 16; off <<= 1)
        v = fmaxf(v, __shfl_xor(v, off, 16));
      const float mnew = fmaxf(rm[j], v);
      const float corr = __expf(rm[j] - mnew);
      rm[j] = mnew;
      const float p0 = __expf(s0[j] - mnew);
      const float p1 = __expf(s1[j] - mnew);
      float rs = p0 + p1;
#pragma unroll
      for (int off = 1; off < 16; off <<= 1)
        rs += __shfl_xor(rs, off, 16);
      rl[j] = rl[j] * corr + rs;
#pragma unroll
      for (int t = 0; t < 4; ++t) o[t][j] *= corr;
      my[(half8 + j) * 32 + row]      = (_Float16)p0;   // P tile -> LDS (row m, col kv)
      my[(half8 + j) * 32 + 16 + row] = (_Float16)p1;
    }

    // C-layout -> A-fragment layout through LDS (same-wave, split DS counter)
    asm volatile("s_wait_dscnt 0" ::: "memory");
    v16h pf;
#pragma unroll
    for (int e = 0; e < 8; ++e) {
      pf[e]     = my[row * 32 + half8 + e];
      pf[e + 8] = my[row * 32 + 16 + half8 + e];
    }

    // ---- O += P(16x32) . V(32x64): B-operand rows from V^T are contiguous ----
#pragma unroll
    for (int t = 0; t < 4; ++t) {
      v16h vfr = load_frag(Vt + (size_t)(t * 16 + row) * S_ + kv0, half8);
      o[t] = __builtin_amdgcn_wmma_f32_16x16x32_f16(false, pf, false, vfr, (short)0, o[t], false, false);
    }
  }

  // ---- normalize, store f16 to [b, s, h, dk] for the output projection ----
  const int b = bh / H_;
  const int h = bh % H_;
#pragma unroll
  for (int j = 0; j < 8; ++j) {
    const float inv = 1.0f / rl[j];
    const int   s   = q0 + half8 + j;
    const size_t base = (((size_t)b * S_ + s) * H_ + h) * DK_;
#pragma unroll
    for (int t = 0; t < 4; ++t)
      Og[base + t * 16 + row] = (_Float16)(o[t][j] * inv);
  }
}

extern "C" void kernel_launch(void* const* d_in, const int* in_sizes, int n_in,
                              void* d_out, int out_size, void* d_ws, size_t ws_size,
                              hipStream_t stream) {
  const float* x      = (const float*)d_in[0];
  const int*   tokpos = (const int*)d_in[1];
  const float* weight = (const float*)d_in[2];
  const float* wo     = (const float*)d_in[3];
  float* out = (float*)d_out;

  char* ws = (char*)d_ws;
  size_t off = 0;
  auto carve = [&](size_t bytes) -> void* {
    void* p = ws + off;
    off += (bytes + 255) & ~(size_t)255;
    return p;
  };

  const size_t nX    = (size_t)B_ * S_ * D_;            // 4M
  const size_t nW    = (size_t)3 * H_ * DK_ * D_;       // 3M
  const size_t nWo   = (size_t)D_ * H_ * DK_;           // 1M
  const size_t nQKV  = (size_t)B_ * S_ * 3 * H_ * DK_;  // 12M
  const size_t nHead = (size_t)B_ * H_ * S_ * DK_;      // 4M

  _Float16* xf16  = (_Float16*)carve(nX * 2);
  _Float16* wf16  = (_Float16*)carve(nW * 2);
  _Float16* wof16 = (_Float16*)carve(nWo * 2);
  float*    qkv32 = (float*)carve(nQKV * 4);
  _Float16* qf16  = (_Float16*)carve(nHead * 2);
  _Float16* kf16  = (_Float16*)carve(nHead * 2);
  _Float16* vtf16 = (_Float16*)carve(nHead * 2);
  _Float16* af16  = (_Float16*)carve(nHead * 2);

  auto nblk = [](size_t n, int t) { return (unsigned)((n + t - 1) / t); };

  f32_to_f16<<<nblk(nX, 256),  256, 0, stream>>>(x, xf16, nX);
  f32_to_f16<<<nblk(nW, 256),  256, 0, stream>>>(weight, wf16, nW);
  f32_to_f16<<<nblk(nWo, 256), 256, 0, stream>>>(wo, wof16, nWo);

  {  // QKV projection: (B*S x D) . (3*H*DK x D)^T -> qkv32
    const int M = B_ * S_, N = 3 * H_ * DK_, K = D_;
    const int tiles = (M / 32) * (N / 64);
    gemm_f16_wmma<<<nblk(tiles, 8), 256, 0, stream>>>(xf16, wf16, qkv32, M, N, K, tiles);
  }

  rope_split<<<nblk((size_t)B_ * H_ * S_ * (DK_ / 2), 256), 256, 0, stream>>>(
      qkv32, tokpos, qf16, kf16, vtf16);

  {  // flash attention: one wave per 16-row query tile
    const int tiles = B_ * H_ * (S_ / 16);  // 4096
    attn_wmma<<<tiles / 8, 256, 0, stream>>>(qf16, kf16, vtf16, af16);
  }

  {  // output projection: (B*S x H*DK) . (D x H*DK)^T -> out
    const int M = B_ * S_, N = D_, K = H_ * DK_;
    const int tiles = (M / 32) * (N / 64);
    gemm_f16_wmma<<<nblk(tiles, 8), 256, 0, stream>>>(af16, wof16, out, M, N, K, tiles);
  }

  (void)in_sizes; (void)n_in; (void)out_size; (void)ws_size;
}
